// DGNLTwo_41291815584273
// MI455X (gfx1250) — matrix operand order repaired and checked by reading, not compiled, not run on hardware
//
#include <hip/hip_runtime.h>
#include <hip/hip_fp16.h>

typedef __attribute__((ext_vector_type(16))) _Float16 v16h;
typedef __attribute__((ext_vector_type(8)))  float    v8f;

#define NB   2
#define CIN  128
#define DCH  64
#define HH   128
#define WW   128
#define HD   64
#define NPIX 4096   // 64*64

#define WMMA_F16(A, B, Cacc) \
  __builtin_amdgcn_wmma_f32_16x16x32_f16(false, (A), false, (B), (short)0, (Cacc), false, false)

// ---------------------------------------------------------------------------
// K1: depthwise 2x2 stride-2 downsample.  xdT: (NB, NPIX, CIN) f16,  df: (NB, NPIX) f32
// ---------------------------------------------------------------------------
__global__ void downsample_kernel(const float* __restrict__ x,
                                  const float* __restrict__ depth,
                                  const float* __restrict__ down_w,
                                  const float* __restrict__ ddown_w,
                                  _Float16* __restrict__ xdT,
                                  float* __restrict__ df) {
  int gid = blockIdx.x * blockDim.x + threadIdx.x;
  if (gid >= NB * NPIX * CIN) return;
  int c = gid & (CIN - 1);
  int m = (gid >> 7) & (NPIX - 1);
  int b = gid >> 19;
  int yy = m >> 6, xx = m & 63;
  const float* xb = x + ((size_t)(b * CIN + c) * HH + 2 * yy) * WW + 2 * xx;
  float w00 = down_w[c * 4 + 0], w01 = down_w[c * 4 + 1];
  float w10 = down_w[c * 4 + 2], w11 = down_w[c * 4 + 3];
  float v = xb[0] * w00 + xb[1] * w01 + xb[WW] * w10 + xb[WW + 1] * w11;
  xdT[((size_t)b * NPIX + m) * CIN + c] = (_Float16)v;
  if (c == 0) {
    const float* db = depth + ((size_t)b * HH + 2 * yy) * WW + 2 * xx;
    df[b * NPIX + m] = db[0] * ddown_w[0] + db[1] * ddown_w[1] +
                       db[WW] * ddown_w[2] + db[WW + 1] * ddown_w[3];
  }
}

// ---------------------------------------------------------------------------
// K2: all six x-projections as one WMMA GEMM.  M=384 rows:
//   [0,64)=phi  [64,128)=f_phi  [128,192)=f_theta  -> QKt (NB,NPIX,192) f16 (transposed)
//   [192,256)=g1 [256,320)=g2 [320,384)=g3         -> Vt  (NB,3,64,NPIX) f16
// grid (64, 24, NB), block 128 (4 waves); wave = one 16x16 D tile.
// ---------------------------------------------------------------------------
__global__ void proj_gemm_kernel(const _Float16* __restrict__ xdT,
    const float* __restrict__ w0, const float* __restrict__ b0,
    const float* __restrict__ w1, const float* __restrict__ b1,
    const float* __restrict__ w2, const float* __restrict__ b2,
    const float* __restrict__ w3, const float* __restrict__ b3,
    const float* __restrict__ w4, const float* __restrict__ b4,
    const float* __restrict__ w5, const float* __restrict__ b5,
    _Float16* __restrict__ QKt, _Float16* __restrict__ Vt) {
  int lane = threadIdx.x & 31;
  int wave = threadIdx.x >> 5;
  int nt = blockIdx.x * 4 + wave;     // 0..255 position tile
  int mt = blockIdx.y;                // 0..23 row tile
  int b  = blockIdx.z;
  int col = lane & 15;
  int hh  = lane >> 4;
  int pos = nt * 16 + col;

  const float* Wm; const float* Bs;
  switch (mt >> 2) {                  // which 64-row weight block (wave-uniform)
    case 0: Wm = w0; Bs = b0; break;
    case 1: Wm = w1; Bs = b1; break;
    case 2: Wm = w2; Bs = b2; break;
    case 3: Wm = w3; Bs = b3; break;
    case 4: Wm = w4; Bs = b4; break;
    default: Wm = w5; Bs = b5; break;
  }
  int arow = (mt & 3) * 16 + col;     // row within the 64-row weight array (A: M = lane&15)

  v8f acc = {};
  for (int kc = 0; kc < 4; ++kc) {    // K = 128 in 4 chunks of 32
    v16h a, bm;
#pragma unroll
    for (int e = 0; e < 16; ++e) {    // A 16x32: K = e+16h (contiguous in e)
      int kk = e + 16 * hh + 32 * kc;
      a[e] = (_Float16)Wm[arow * CIN + kk];
    }
    const _Float16* bp = xdT + ((size_t)b * NPIX + pos) * CIN + 16 * hh + 32 * kc;
#pragma unroll
    for (int e = 0; e < 16; ++e)      // B 32x16: K = e+16h (contiguous in e)
      bm[e] = bp[e];
    acc = WMMA_F16(a, bm, acc);
  }
#pragma unroll
  for (int r = 0; r < 8; ++r) {       // D: M = r+8h, N = col
    int row = mt * 16 + r + 8 * hh;
    float v = acc[r] + Bs[row & 63];
    if (row < 192) {
      QKt[((size_t)b * NPIX + pos) * 192 + row] = (_Float16)v;
    } else {
      int gi = (row - 192) >> 6;
      int j  = row & 63;
      Vt[(((size_t)b * 3 + gi) * DCH + j) * NPIX + pos] = (_Float16)v;
    }
  }
}

// ---------------------------------------------------------------------------
// K3: flash attention.  One wave = 32 query rows (two 16-row A-tiles sharing
// every K/V B-operand -> 16 WMMAs per 32-key block, ~32 flops per L2 byte),
// streaming all 4096 keys with online softmax.
//   attn 0: Q=theta(df), K=phi,      V=g1
//   attn 1: Q=d_theta(df), K=d_phi(df), V=g2
//   attn 2: Q=f_theta,   K=f_phi,    V=g3
// grid (64, 3, NB), block 64 (2 waves).
// ---------------------------------------------------------------------------
__global__ void flash_attn_kernel(const _Float16* __restrict__ QKt,
                                  const _Float16* __restrict__ Vt,
                                  const float* __restrict__ df,
                                  const float* __restrict__ theta_w, const float* __restrict__ theta_b,
                                  const float* __restrict__ dtheta_w, const float* __restrict__ dtheta_b,
                                  const float* __restrict__ dphi_w,  const float* __restrict__ dphi_b,
                                  float* __restrict__ Oout) {
  __shared__ _Float16 plds[2][2][16 * 32];  // [wave][q-subtile] P-transpose staging
  int lane = threadIdx.x & 31;
  int wave = threadIdx.x >> 5;
  int qt   = blockIdx.x * 2 + wave;         // 0..127 (32 rows each)
  int attn = blockIdx.y;
  int b    = blockIdx.z;
  int col  = lane & 15;
  int hh   = lane >> 4;
  int qbase = qt * 32;
  const float* dfb = df + b * NPIX;

  // ---- Q in A layout: two 16-row sub-tiles x two K=32 chunks ----
  v16h aQ[2][2];
#pragma unroll
  for (int u = 0; u < 2; ++u) {
    int qrow = qbase + u * 16 + col;
    float dfq = dfb[qrow];
#pragma unroll
    for (int e = 0; e < 16; ++e) {
      int j0 = e + 16 * hh;
      int j1 = j0 + 32;
      float q0, q1;
      if (attn == 0)      { q0 = theta_w[j0] * dfq + theta_b[j0];   q1 = theta_w[j1] * dfq + theta_b[j1]; }
      else if (attn == 1) { q0 = dtheta_w[j0] * dfq + dtheta_b[j0]; q1 = dtheta_w[j1] * dfq + dtheta_b[j1]; }
      else {
        const _Float16* qp = QKt + ((size_t)b * NPIX + qrow) * 192 + 128;
        q0 = (float)qp[j0]; q1 = (float)qp[j1];
      }
      aQ[u][0][e] = (_Float16)q0; aQ[u][1][e] = (_Float16)q1;
    }
  }

  v8f o[2][4] = {};
  float mrun[2][8], lrun[2][8];
#pragma unroll
  for (int u = 0; u < 2; ++u)
#pragma unroll
    for (int r = 0; r < 8; ++r) { mrun[u][r] = -1e30f; lrun[u][r] = 0.0f; }

  const _Float16* vbase = Vt + (((size_t)b * 3 + attn) * DCH) * NPIX;
  int koff = (attn == 0) ? 0 : 64;          // phi vs f_phi column block

  for (int kt = 0; kt < NPIX / 32; ++kt) {
    int key0 = kt * 32;
    // prefetch next block's K rows and V rows (streaming, predictable)
    if (kt + 1 < NPIX / 32) {
      int keyn = key0 + 32;
      if (attn != 1)
        __builtin_prefetch(QKt + ((size_t)b * NPIX + keyn + lane) * 192 + koff, 0, 1);
      __builtin_prefetch(vbase + (size_t)lane * NPIX + keyn, 0, 1);
      __builtin_prefetch(vbase + (size_t)(lane + 32) * NPIX + keyn, 0, 1);
    }
    // ---- K operand, B layout, both 16-key sub-tiles (shared by both Q tiles) ----
    v16h bKa0, bKa1, bKb0, bKb1;
    if (attn == 1) {
      float dfk0 = dfb[key0 + col];
      float dfk1 = dfb[key0 + 16 + col];
#pragma unroll
      for (int e = 0; e < 16; ++e) {
        int j0 = e + 16 * hh, j1 = j0 + 32;
        float wv0 = dphi_w[j0], bv0 = dphi_b[j0];
        float wv1 = dphi_w[j1], bv1 = dphi_b[j1];
        bKa0[e] = (_Float16)(wv0 * dfk0 + bv0);
        bKa1[e] = (_Float16)(wv1 * dfk0 + bv1);
        bKb0[e] = (_Float16)(wv0 * dfk1 + bv0);
        bKb1[e] = (_Float16)(wv1 * dfk1 + bv1);
      }
    } else {
      const _Float16* kp0 = QKt + ((size_t)b * NPIX + key0 + col) * 192 + koff + 16 * hh;
      const _Float16* kp1 = kp0 + (size_t)16 * 192;
#pragma unroll
      for (int e = 0; e < 16; ++e) {
        bKa0[e] = kp0[e];  bKa1[e] = kp0[e + 32];
        bKb0[e] = kp1[e];  bKb1[e] = kp1[e + 32];
      }
    }
    // ---- per Q sub-tile: scores + online softmax + P staging ----
#pragma unroll
    for (int u = 0; u < 2; ++u) {
      v8f sA = {}, sB = {};
      sA = WMMA_F16(aQ[u][0], bKa0, sA);
      sA = WMMA_F16(aQ[u][1], bKa1, sA);
      sB = WMMA_F16(aQ[u][0], bKb0, sB);
      sB = WMMA_F16(aQ[u][1], bKb1, sB);
      _Float16* pl = plds[wave][u];
#pragma unroll
      for (int r = 0; r < 8; ++r) {
        float t = fmaxf(sA[r], sB[r]);
        t = fmaxf(t, __shfl_xor(t, 1, 16));
        t = fmaxf(t, __shfl_xor(t, 2, 16));
        t = fmaxf(t, __shfl_xor(t, 4, 16));
        t = fmaxf(t, __shfl_xor(t, 8, 16));
        float newm = fmaxf(mrun[u][r], t);
        float pA = __expf(sA[r] - newm);
        float pB = __expf(sB[r] - newm);
        float ps = pA + pB;
        ps += __shfl_xor(ps, 1, 16);
        ps += __shfl_xor(ps, 2, 16);
        ps += __shfl_xor(ps, 4, 16);
        ps += __shfl_xor(ps, 8, 16);
        float scale = __expf(mrun[u][r] - newm);
        lrun[u][r] = lrun[u][r] * scale + ps;
        mrun[u][r] = newm;
        o[u][0][r] *= scale; o[u][1][r] *= scale;
        o[u][2][r] *= scale; o[u][3][r] *= scale;
        pl[(r + 8 * hh) * 32 + col]      = (_Float16)pA;  // C layout -> LDS [qrow][key]
        pl[(r + 8 * hh) * 32 + 16 + col] = (_Float16)pB;
      }
    }
    // ---- P back in A layout for both sub-tiles ----
    v16h aP0, aP1;
#pragma unroll
    for (int e = 0; e < 16; ++e) {
      int off = e + 16 * hh;
      aP0[e] = plds[wave][0][col * 32 + off];
      aP1[e] = plds[wave][1][col * 32 + off];
    }
    // ---- O += P @ V: each V B-tile feeds both Q sub-tiles ----
#pragma unroll
    for (int jt = 0; jt < 4; ++jt) {
      v16h bV;
      const _Float16* vp = vbase + (size_t)(jt * 16 + col) * NPIX + key0 + 16 * hh;
#pragma unroll
      for (int e = 0; e < 16; ++e) bV[e] = vp[e];
      o[0][jt] = WMMA_F16(aP0, bV, o[0][jt]);
      o[1][jt] = WMMA_F16(aP1, bV, o[1][jt]);
    }
  }
  // ---- normalize + store O (NB,3,NPIX,64) f32 ----
  float* ob = Oout + (((size_t)b * 3 + attn) * NPIX) * DCH;
#pragma unroll
  for (int u = 0; u < 2; ++u) {
#pragma unroll
    for (int r = 0; r < 8; ++r) {
      float inv = 1.0f / lrun[u][r];
      size_t row = (size_t)(qbase + u * 16 + r + 8 * hh) * DCH;
      ob[row + 0 + col]  = o[u][0][r] * inv;
      ob[row + 16 + col] = o[u][1][r] * inv;
      ob[row + 32 + col] = o[u][2][r] * inv;
      ob[row + 48 + col] = o[u][3][r] * inv;
    }
  }
}

// ---------------------------------------------------------------------------
// K4: zf = z_w @ (O0+O1+O2)^T + z_b   (M=128, K=64, N=4096 per batch)
// grid (128, 8, NB), block 64.
// ---------------------------------------------------------------------------
__global__ void zconv_gemm_kernel(const float* __restrict__ Oout,
                                  const float* __restrict__ z_w,
                                  const float* __restrict__ z_b,
                                  float* __restrict__ zf) {
  int lane = threadIdx.x & 31;
  int wave = threadIdx.x >> 5;
  int nt = blockIdx.x * 2 + wave;
  int mt = blockIdx.y;
  int b  = blockIdx.z;
  int col = lane & 15;
  int hh  = lane >> 4;
  int pos = nt * 16 + col;
  const float* ob0 = Oout + (((size_t)b * 3 + 0) * NPIX) * DCH;
  const float* ob1 = Oout + (((size_t)b * 3 + 1) * NPIX) * DCH;
  const float* ob2 = Oout + (((size_t)b * 3 + 2) * NPIX) * DCH;

  v8f acc = {};
#pragma unroll
  for (int kc = 0; kc < 2; ++kc) {
    v16h a, bm;
#pragma unroll
    for (int e = 0; e < 16; ++e) {
      int kk = e + 16 * hh + 32 * kc;
      a[e] = (_Float16)z_w[(mt * 16 + col) * DCH + kk];
    }
    size_t base = (size_t)pos * DCH + 16 * hh + 32 * kc;
#pragma unroll
    for (int e = 0; e < 16; ++e)
      bm[e] = (_Float16)(ob0[base + e] + ob1[base + e] + ob2[base + e]);
    acc = WMMA_F16(a, bm, acc);
  }
#pragma unroll
  for (int r = 0; r < 8; ++r) {
    int co = mt * 16 + r + 8 * hh;
    zf[((size_t)b * CIN + co) * NPIX + pos] = acc[r] + z_b[co];
  }
}

// ---------------------------------------------------------------------------
// K5: out = x + bilinear_align_corners(zf, 128, 128)
// ---------------------------------------------------------------------------
__global__ void upsample_add_kernel(const float* __restrict__ x,
                                    const float* __restrict__ zf,
                                    float* __restrict__ out) {
  int gid = blockIdx.x * blockDim.x + threadIdx.x;
  if (gid >= NB * CIN * HH * WW) return;
  int xo = gid & 127;
  int yo = (gid >> 7) & 127;
  int cb = gid >> 14;   // b*CIN + co
  const float scl = 63.0f / 127.0f;
  float ys = yo * scl, xs = xo * scl;
  int y0 = (int)floorf(ys); int y1 = min(y0 + 1, 63);
  int x0 = (int)floorf(xs); int x1 = min(x0 + 1, 63);
  float wy = ys - (float)y0, wx = xs - (float)x0;
  const float* zb = zf + (size_t)cb * NPIX;
  float v00 = zb[y0 * HD + x0], v01 = zb[y0 * HD + x1];
  float v10 = zb[y1 * HD + x0], v11 = zb[y1 * HD + x1];
  float top = v00 * (1.0f - wx) + v01 * wx;
  float bot = v10 * (1.0f - wx) + v11 * wx;
  out[gid] = x[gid] + top * (1.0f - wy) + bot * wy;
}

// ---------------------------------------------------------------------------
extern "C" void kernel_launch(void* const* d_in, const int* in_sizes, int n_in,
                              void* d_out, int out_size, void* d_ws, size_t ws_size,
                              hipStream_t stream) {
  (void)in_sizes; (void)n_in; (void)out_size; (void)ws_size;
  const float* x          = (const float*)d_in[0];
  const float* depth_map  = (const float*)d_in[1];
  const float* down_w     = (const float*)d_in[2];
  const float* ddown_w    = (const float*)d_in[3];
  const float* phi_w      = (const float*)d_in[4];
  const float* phi_b      = (const float*)d_in[5];
  const float* f_phi_w    = (const float*)d_in[6];
  const float* f_phi_b    = (const float*)d_in[7];
  const float* f_theta_w  = (const float*)d_in[8];
  const float* f_theta_b  = (const float*)d_in[9];
  const float* g1_w       = (const float*)d_in[10];
  const float* g1_b       = (const float*)d_in[11];
  const float* g2_w       = (const float*)d_in[12];
  const float* g2_b       = (const float*)d_in[13];
  const float* g3_w       = (const float*)d_in[14];
  const float* g3_b       = (const float*)d_in[15];
  const float* theta_w    = (const float*)d_in[16];
  const float* theta_b    = (const float*)d_in[17];
  const float* d_theta_w  = (const float*)d_in[18];
  const float* d_theta_b  = (const float*)d_in[19];
  const float* d_phi_w    = (const float*)d_in[20];
  const float* d_phi_b    = (const float*)d_in[21];
  const float* z_w        = (const float*)d_in[22];
  const float* z_b        = (const float*)d_in[23];

  // workspace partition (~18 MB)
  _Float16* xdT = (_Float16*)d_ws;                       // NB*NPIX*CIN
  float*    df  = (float*)(xdT + (size_t)NB * NPIX * CIN);
  _Float16* QKt = (_Float16*)(df + (size_t)NB * NPIX);   // NB*NPIX*192
  _Float16* Vt  = QKt + (size_t)NB * NPIX * 192;         // NB*3*64*NPIX
  float*    Oo  = (float*)(Vt + (size_t)NB * 3 * DCH * NPIX); // NB*3*NPIX*64
  float*    zf  = Oo + (size_t)NB * 3 * NPIX * DCH;      // NB*CIN*NPIX

  downsample_kernel<<<dim3((NB * NPIX * CIN + 255) / 256), 256, 0, stream>>>(
      x, depth_map, down_w, ddown_w, xdT, df);

  proj_gemm_kernel<<<dim3(64, 24, NB), 128, 0, stream>>>(
      xdT, phi_w, phi_b, f_phi_w, f_phi_b, f_theta_w, f_theta_b,
      g1_w, g1_b, g2_w, g2_b, g3_w, g3_b, QKt, Vt);

  flash_attn_kernel<<<dim3(64, 3, NB), 64, 0, stream>>>(
      QKt, Vt, df, theta_w, theta_b, d_theta_w, d_theta_b, d_phi_w, d_phi_b, Oo);

  zconv_gemm_kernel<<<dim3(128, 8, NB), 64, 0, stream>>>(Oo, z_w, z_b, zf);

  upsample_add_kernel<<<dim3((NB * CIN * HH * WW + 255) / 256), 256, 0, stream>>>(
      x, zf, (float*)d_out);
}
